// Attention_56745107915012
// MI455X (gfx1250) — matrix operand order repaired and checked by reading, not compiled, not run on hardware
//
#include <hip/hip_runtime.h>
#include <hip/hip_bf16.h>

// ---------------------------------------------------------------------------
// Attention block for MI455X (gfx1250, wave32, WMMA f16->f32).
//   hidden[2,2048,1024] f32, w_qkv[3072,1024], b_qkv[3072], w_o[1024,1024],
//   b_o[1024] -> out[2,2048,1024] f32.
// f16 WMMA with f32 accumulate; f32 softmax. GEMM K-loops are FULLY UNROLLED
// (trip 32) so every load is base + immediate offset: no per-iteration
// address VALU, no WMMA hazard NOPs, no ping-pong live ranges to spill.
// ---------------------------------------------------------------------------

typedef _Float16 half_t;
typedef __attribute__((ext_vector_type(16))) _Float16 v16h;
typedef __attribute__((ext_vector_type(8)))  _Float16 v8h;
typedef __attribute__((ext_vector_type(4)))  _Float16 v4h;
typedef __attribute__((ext_vector_type(8)))  float    v8f;
typedef __attribute__((ext_vector_type(4)))  float    v4f;

#define NHEADS   16
#define HEAD_DIM 64
#define HIDDEN   1024
#define QKV      1024
#define SEQ      2048
#define BATCH    2
#define ROWS     (BATCH * SEQ)          // 4096
#define SCALING  0.125f                 // 64^-0.5

// ---- WMMA fragment loaders -------------------------------------------------
// A matrix 16x32 f16: row m=lane&15; halves e=0..15 map to
// K = e + 16*(lane>=16) (contiguous). base = row ptr + (lane>=16)*16.
__device__ __forceinline__ v16h frag_a(const half_t* base) {
    v8h lo = *(const v8h*)base;
    v8h hi = *(const v8h*)(base + 8);
    v16h r;
#pragma unroll
    for (int i = 0; i < 8; ++i) { r[i] = lo[i]; r[8 + i] = hi[i]; }
    return r;
}

// B matrix 32x16 f16: lane holds column n = lane&15; halves e=0..15 map to
// K = e + 16*(lane>=16) (contiguous). base = source-row ptr + (lane>=16)*16.
__device__ __forceinline__ v16h frag_b(const half_t* base) {
    v8h lo = *(const v8h*)base;
    v8h hi = *(const v8h*)(base + 8);
    v16h r;
#pragma unroll
    for (int i = 0; i < 8; ++i) { r[i] = lo[i]; r[8 + i] = hi[i]; }
    return r;
}

#define WMMA_F16(A, B, C) \
    __builtin_amdgcn_wmma_f32_16x16x32_f16(false, (A), false, (B), (short)0, (C), false, false)

// ---- vectorized f32 -> f16 cast --------------------------------------------
__global__ __launch_bounds__(256) void cast_f16(const float* __restrict__ in,
                                                half_t* __restrict__ out, int n4) {
    int i = blockIdx.x * 256 + threadIdx.x;
    if (i < n4) {
        v4f x = ((const v4f*)in)[i];
        v4h y;
#pragma unroll
        for (int k = 0; k < 4; ++k) y[k] = (half_t)x[k];
        ((v4h*)out)[i] = y;
    }
}

// ---- shared 32x64 wave-tile GEMM core (K = 1024), fully unrolled -----------
// Au/Bu are tile-uniform base pointers; aoff0/aoff1/boff are lane offsets.
// Full unroll: all k indices are compile-time constants -> immediate offsets.
__device__ __forceinline__ void gemm_tile_32x64(const half_t* __restrict__ Au,
                                                const half_t* __restrict__ Bu,
                                                int aoff0, int aoff1, int boff,
                                                v8f acc[2][4]) {
    v16h a[2][2], b[2][4];
    a[0][0] = frag_a(Au + aoff0);
    a[0][1] = frag_a(Au + aoff1);
#pragma unroll
    for (int j = 0; j < 4; ++j) b[0][j] = frag_b(Bu + boff + j * (16 * 1024));

#pragma unroll
    for (int it = 0; it < 32; ++it) {
        const int buf = it & 1;
        const int nk  = ((it + 1) & 31) * 32;   // wrapped, unconditional prefetch
        a[buf ^ 1][0] = frag_a(Au + nk + aoff0);
        a[buf ^ 1][1] = frag_a(Au + nk + aoff1);
#pragma unroll
        for (int j = 0; j < 4; ++j)
            b[buf ^ 1][j] = frag_b(Bu + nk + boff + j * (16 * 1024));
#pragma unroll
        for (int j = 0; j < 4; ++j)
#pragma unroll
            for (int i = 0; i < 2; ++i)
                acc[i][j] = WMMA_F16(a[buf][i], b[buf][j], acc[i][j]);
    }
}

// ---- QKV projection: X[4096,1024] x Wqkv^T + b -> Q,K ([b,h,s,d]), Vt ([b,h,d,s])
__global__ __launch_bounds__(256) void qkv_gemm(const half_t* __restrict__ X,
                                                const half_t* __restrict__ W,
                                                const float* __restrict__ bias,
                                                half_t* __restrict__ Q,
                                                half_t* __restrict__ Kd,
                                                half_t* __restrict__ Vt) {
    const int lane = threadIdx.x & 31;
    const int wave = threadIdx.x >> 5;
    const int tile = blockIdx.x * 8 + wave;      // 128 Mtiles(32) * 48 Ngroups(64)
    const int mt = tile / 48, ng = tile % 48;
    const int m0 = mt * 32, n0 = ng * 64;
    const int hi = (lane >> 4) & 1;
    const int lm = lane & 15;

    const int aoff = lm * HIDDEN + hi * 16;
    const int boff = lm * HIDDEN + hi * 16;
    v8f acc[2][4] = {};
    gemm_tile_32x64(X + (size_t)m0 * HIDDEN, W + (size_t)n0 * HIDDEN,
                    aoff, aoff + 16 * HIDDEN, boff, acc);

    // Tile-uniform decomposition: n0 multiple of 64 -> one section/head per
    // tile; m0 multiple of 32 -> one batch per tile.
    const int sect = n0 >> 10;                   // 0=Q 1=K 2=V
    const int h    = (n0 & (QKV - 1)) >> 6;
    const int b_   = m0 >> 11;
    const int s0   = m0 & (SEQ - 1);
    const size_t bhs = (size_t)b_ * NHEADS + h;
    float bv[4];
#pragma unroll
    for (int j = 0; j < 4; ++j) bv[j] = bias[n0 + j * 16 + lm];

    if (sect < 2) {                              // Q or K: [bh, s, d] rows
        half_t* dst = (sect == 0 ? Q : Kd) + (bhs * SEQ + s0) * HEAD_DIM;
#pragma unroll
        for (int j = 0; j < 4; ++j)
#pragma unroll
            for (int i = 0; i < 2; ++i)
#pragma unroll
                for (int v = 0; v < 8; ++v)
                    dst[(i * 16 + v + hi * 8) * HEAD_DIM + j * 16 + lm] =
                        (half_t)(acc[i][j][v] + bv[j]);
    } else {                                     // V transposed: [bh, d, s]
        half_t* dst = Vt + bhs * HEAD_DIM * SEQ + s0;
#pragma unroll
        for (int j = 0; j < 4; ++j)
#pragma unroll
            for (int i = 0; i < 2; ++i)
#pragma unroll
                for (int v = 0; v < 8; ++v)
                    dst[(j * 16 + lm) * SEQ + i * 16 + v + hi * 8] =
                        (half_t)(acc[i][j][v] + bv[j]);
    }
}

// ---- Flash attention: one wave = one (b,h, 16-query) tile ------------------
__global__ __launch_bounds__(256) void attn_kernel(const half_t* __restrict__ Q,
                                                   const half_t* __restrict__ Kd,
                                                   const half_t* __restrict__ Vt,
                                                   half_t* __restrict__ Aout) {
    __shared__ __align__(16) half_t Pst[8][16 * 32];   // per-wave P staging
    const int lane = threadIdx.x & 31;
    const int wave = threadIdx.x >> 5;
    const int tile = blockIdx.x * 8 + wave;            // 32 bh * 128 qtiles
    const int bh = tile >> 7, qt = tile & 127;
    const int hi = (lane >> 4) & 1;
    const int lm = lane & 15;

    const half_t* Qb = Q  + ((size_t)bh * SEQ + qt * 16) * HEAD_DIM;  // uniform
    const half_t* Kb = Kd + (size_t)bh * SEQ * HEAD_DIM;              // uniform
    const half_t* Vb = Vt + (size_t)bh * HEAD_DIM * SEQ;              // uniform
    half_t* Pw = &Pst[wave][0];

    const int qoff = lm * HEAD_DIM + hi * 16;      // lane offsets (32-bit)
    const int koff = lm * HEAD_DIM + hi * 16;
    const int voff = lm * SEQ + hi * 16;

    const v16h aQ0 = frag_a(Qb + qoff);            // d 0..31
    const v16h aQ1 = frag_a(Qb + qoff + 32);       // d 32..63

    float mrow[8], lrow[8];
    v8f accO[4] = {};
#pragma unroll
    for (int v = 0; v < 8; ++v) { mrow[v] = -1e30f; lrow[v] = 0.0f; }

    // prime K fragments for chunk 0: bk[buf][key-half nh][d-half]
    v16h bk[2][2][2];
#pragma unroll
    for (int nh = 0; nh < 2; ++nh) {
        bk[0][nh][0] = frag_b(Kb + nh * (16 * HEAD_DIM) + koff);
        bk[0][nh][1] = frag_b(Kb + nh * (16 * HEAD_DIM) + koff + 32);
    }

#pragma unroll 2
    for (int kc = 0, buf = 0; kc < SEQ; kc += 32, buf ^= 1) {
        const int knext = (kc + 32) & (SEQ - 1);   // wrapped, branch-free
        const int krow  = knext * HEAD_DIM;        // uniform k-advance

        // V fragments for this chunk -- issue loads before the score WMMAs
        v16h bV[4];
#pragma unroll
        for (int j = 0; j < 4; ++j)
            bV[j] = frag_b(Vb + kc + j * (16 * SEQ) + voff);
        __builtin_prefetch(Vb + knext + voff, 0, 1);

        // scores 16q x 32k, accumulated over D=64
        v8f sc[2];
#pragma unroll
        for (int nh = 0; nh < 2; ++nh) {
            v8f s = {};
            s = WMMA_F16(aQ0, bk[buf][nh][0], s);
            s = WMMA_F16(aQ1, bk[buf][nh][1], s);
            sc[nh] = s;
        }
        // prefetch next chunk's K fragments into the other buffer (wrapped)
#pragma unroll
        for (int nh = 0; nh < 2; ++nh) {
            bk[buf ^ 1][nh][0] = frag_b(Kb + krow + nh * (16 * HEAD_DIM) + koff);
            bk[buf ^ 1][nh][1] = frag_b(Kb + krow + nh * (16 * HEAD_DIM) + koff + 32);
        }

        // online softmax (f32) -- VALU work overlaps the loads in flight
        float scale_[8];
#pragma unroll
        for (int v = 0; v < 8; ++v) {
            float x0 = sc[0][v] * SCALING, x1 = sc[1][v] * SCALING;
            sc[0][v] = x0; sc[1][v] = x1;
            float mx = fmaxf(x0, x1);
#pragma unroll
            for (int msk = 1; msk < 16; msk <<= 1)
                mx = fmaxf(mx, __shfl_xor(mx, msk, 32));
            const float nm = fmaxf(mrow[v], mx);
            scale_[v] = __expf(mrow[v] - nm);
            mrow[v] = nm;
            lrow[v] *= scale_[v];
        }
#pragma unroll
        for (int j = 0; j < 4; ++j)
#pragma unroll
            for (int v = 0; v < 8; ++v) accO[j][v] *= scale_[v];
#pragma unroll
        for (int v = 0; v < 8; ++v) {
            const float p0 = __expf(sc[0][v] - mrow[v]);
            const float p1 = __expf(sc[1][v] - mrow[v]);
            float rs = p0 + p1;
#pragma unroll
            for (int msk = 1; msk < 16; msk <<= 1)
                rs += __shfl_xor(rs, msk, 32);
            lrow[v] += rs;
            // C-layout (M = v + hi*8, N = lm) -> LDS row-major P[16][32]
            Pw[(v + hi * 8) * 32 + lm]      = (half_t)p0;
            Pw[(v + hi * 8) * 32 + 16 + lm] = (half_t)p1;
        }
        // reload P in A layout (LDS is in-order per wave) and do P x V
        const v16h aP = frag_a(Pw + lm * 32 + hi * 16);
#pragma unroll
        for (int j = 0; j < 4; ++j)
            accO[j] = WMMA_F16(aP, bV[j], accO[j]);
    }

    // epilogue: normalize and write [b, s, h*64 + d] in f16
    const int b_ = bh >> 4, h = bh & 15;
    half_t* dst = Aout + ((size_t)b_ * SEQ + qt * 16) * QKV + h * HEAD_DIM;
#pragma unroll
    for (int j = 0; j < 4; ++j)
#pragma unroll
        for (int v = 0; v < 8; ++v)
            dst[(v + hi * 8) * QKV + j * 16 + lm] =
                (half_t)(accO[j][v] / lrow[v]);
}

// ---- Output projection: Aout[4096,1024] x Wo^T + b_o -> f32 out ------------
__global__ __launch_bounds__(256) void oproj_gemm(const half_t* __restrict__ X,
                                                  const half_t* __restrict__ W,
                                                  const float* __restrict__ bias,
                                                  float* __restrict__ out) {
    const int lane = threadIdx.x & 31;
    const int wave = threadIdx.x >> 5;
    const int tile = blockIdx.x * 8 + wave;      // 128 Mtiles(32) * 16 Ngroups(64)
    const int mt = tile / 16, ng = tile % 16;
    const int m0 = mt * 32, n0 = ng * 64;
    const int hi = (lane >> 4) & 1;
    const int lm = lane & 15;

    const int aoff = lm * QKV + hi * 16;
    const int boff = lm * QKV + hi * 16;
    v8f acc[2][4] = {};
    gemm_tile_32x64(X + (size_t)m0 * QKV, W + (size_t)n0 * QKV,
                    aoff, aoff + 16 * QKV, boff, acc);

    float* dst = out + (size_t)m0 * HIDDEN + n0 + lm;
#pragma unroll
    for (int j = 0; j < 4; ++j) {
        const float bv = bias[n0 + j * 16 + lm];
#pragma unroll
        for (int i = 0; i < 2; ++i)
#pragma unroll
            for (int v = 0; v < 8; ++v)
                dst[(i * 16 + v + hi * 8) * HIDDEN + j * 16] = acc[i][j][v] + bv;
    }
}

// ---------------------------------------------------------------------------
extern "C" void kernel_launch(void* const* d_in, const int* in_sizes, int n_in,
                              void* d_out, int out_size, void* d_ws, size_t ws_size,
                              hipStream_t stream) {
    const float* hs     = (const float*)d_in[0];   // [2,2048,1024]
    const float* w_qkv  = (const float*)d_in[1];   // [3072,1024]
    const float* b_qkv  = (const float*)d_in[2];   // [3072]
    const float* w_o    = (const float*)d_in[3];   // [1024,1024]
    const float* b_o    = (const float*)d_in[4];   // [1024]
    float* out = (float*)d_out;

    char* ws = (char*)d_ws;
    half_t* hX  = (half_t*)ws;  ws += (size_t)ROWS * HIDDEN * 2;        // 8 MB
    half_t* hWq = (half_t*)ws;  ws += (size_t)3 * QKV * HIDDEN * 2;     // 6 MB
    half_t* hWo = (half_t*)ws;  ws += (size_t)HIDDEN * QKV * 2;         // 2 MB
    half_t* Qh  = (half_t*)ws;  ws += (size_t)ROWS * QKV * 2;           // 8 MB
    half_t* Kh  = (half_t*)ws;  ws += (size_t)ROWS * QKV * 2;           // 8 MB
    half_t* Vt  = (half_t*)ws;  ws += (size_t)ROWS * QKV * 2;           // 8 MB
    half_t* Ao  = (half_t*)ws;  ws += (size_t)ROWS * QKV * 2;           // 8 MB

    const int nX  = ROWS * HIDDEN;        // 4,194,304
    const int nWq = 3 * QKV * HIDDEN;     // 3,145,728
    const int nWo = HIDDEN * QKV;         // 1,048,576
    cast_f16<<<(nX  / 4 + 255) / 256, 256, 0, stream>>>(hs,    hX,  nX  / 4);
    cast_f16<<<(nWq / 4 + 255) / 256, 256, 0, stream>>>(w_qkv, hWq, nWq / 4);
    cast_f16<<<(nWo / 4 + 255) / 256, 256, 0, stream>>>(w_o,   hWo, nWo / 4);

    // 128 Mtiles * 48 Ngroups = 6144 wave-tiles / 8 waves per block
    qkv_gemm<<<768, 256, 0, stream>>>(hX, hWq, b_qkv, Qh, Kh, Vt);
    // 32 (b,h) * 128 qtiles = 4096 wave-tiles / 8
    attn_kernel<<<512, 256, 0, stream>>>(Qh, Kh, Vt, Ao);
    // 128 Mtiles * 16 Ngroups = 2048 wave-tiles / 8
    oproj_gemm<<<256, 256, 0, stream>>>(Ao, hWo, b_o, out);
}
